// GNNEncoders_8581344657807
// MI455X (gfx1250) — matrix operand order repaired and checked by reading, not compiled, not run on hardware
//
#include <hip/hip_runtime.h>
#include <hip/hip_bf16.h>

#define NN   50000
#define DHH  64
#define HEADS 4
#define HD   256          // HEADS*DHH
#define ETOT 900000
#define WPAD 72           // padded LDS row stride (halves) for bank spread

typedef unsigned short u16;
typedef __attribute__((ext_vector_type(16))) __bf16 v16bf;
typedef __attribute__((ext_vector_type(8)))  float  v8f;

__device__ __forceinline__ u16 f2bf(float f) {
    unsigned u = __float_as_uint(f);
    unsigned r = u + 0x7fffu + ((u >> 16) & 1u);   // RNE
    return (u16)(r >> 16);
}
// pack two floats -> two bf16 in one dword (v_cvt_pk_bf16_f32 when available)
__device__ __forceinline__ unsigned f2bf2(float a, float b) {
#if __has_builtin(__builtin_amdgcn_cvt_pk_bf16_f32)
    auto pk = __builtin_amdgcn_cvt_pk_bf16_f32(a, b);
    unsigned u; __builtin_memcpy(&u, &pk, 4);
    return u;
#else
    return (unsigned)f2bf(a) | ((unsigned)f2bf(b) << 16);
#endif
}
__device__ __forceinline__ float bf2f(u16 h) { return __uint_as_float(((unsigned)h) << 16); }

// monotone float<->int key for integer atomicMax
__device__ __forceinline__ int   f2ord(float f) { int b = __float_as_int(f); return b >= 0 ? b : (b ^ 0x7fffffff); }
__device__ __forceinline__ float ord2f(int k)   { int b = k >= 0 ? k : (k ^ 0x7fffffff); return __int_as_float(b); }

// ---------------- BatchNorm ----------------
__global__ void zero_stats(float* stats) {
    int t = threadIdx.x;
    if (t < 128) stats[t] = 0.f;   // sum[0:64], sumsq[64:128]
}

__global__ void bn_stats(const float* __restrict__ x, float* __restrict__ sum,
                         float* __restrict__ sumsq) {
    int tid = blockIdx.x * 256 + threadIdx.x;   // 65536 threads
    int d = tid & 63;
    int slice = tid >> 6;                        // 0..1023
    float s = 0.f, q = 0.f;
    for (int r = slice; r < NN; r += 1024) {
        float v = x[(size_t)r * 64 + d];
        s += v; q += v * v;
    }
    atomicAdd(&sum[d], s);
    atomicAdd(&sumsq[d], q);
}

__global__ void bn_apply(const float* __restrict__ x, const float* __restrict__ sum,
                         const float* __restrict__ sumsq, const float* __restrict__ g,
                         const float* __restrict__ be, u16* __restrict__ xn) {
    int gid = blockIdx.x * 256 + threadIdx.x;
    if (gid >= NN * 64) return;
    int d = gid & 63;
    float mu  = sum[d] * (1.0f / NN);
    float var = sumsq[d] * (1.0f / NN) - mu * mu;
    float sc  = g[d] * rsqrtf(var + 1e-5f);
    xn[gid] = f2bf((x[gid] - mu) * sc + be[d]);
}

// ---------------- weight prep: fp32 -> bf16, residual pre-sum ----------------
__global__ void prep_weights(const float* __restrict__ W, const float* __restrict__ rW,
                             const float* __restrict__ b, const float* __restrict__ fcW,
                             u16* __restrict__ Wbf, u16* __restrict__ Wres,
                             float* __restrict__ bsum, u16* __restrict__ fcWbf) {
    int i = blockIdx.x * 256 + threadIdx.x;      // 49152 threads
    if (i < 3 * 64 * HD) Wbf[i] = f2bf(W[i]);
    if (i < 64 * HD)     Wres[i] = f2bf(rW[i] + rW[i + 64 * HD] + rW[i + 2 * 64 * HD]);
    if (i < 64 * 64)     fcWbf[i] = f2bf(fcW[i]);
    if (i < HD)          bsum[i] = b[i] + b[i + HD] + b[i + 2 * HD];
}

// ---------------- WMMA GEMM: C[NN,NC] = A[NN,64](bf16) x B[64,NC](bf16) (+bias) ----------------
__device__ __forceinline__ v16bf loadA(const u16* __restrict__ Arow, int k0, int hi) {
    // A 16x32 layout (contiguous, matches B): lane half i -> K = k0 + hi*16 + i
    uint4 a = *(const uint4*)(Arow + k0 + hi * 16);       // K = k0+hi*16 .. +7
    uint4 b = *(const uint4*)(Arow + k0 + hi * 16 + 8);   // K = k0+hi*16+8 .. +15
    union { uint4 q[2]; v16bf v; } u;
    u.q[0] = a; u.q[1] = b;
    return u.v;
}

__device__ __forceinline__ v16bf loadBt(const u16* __restrict__ ldsWt, int n, int k0, int hi) {
    // B 32x16 layout: lane holds column n; half i -> K = k0 + hi*16 + i (contiguous in [n][k] LDS)
    const u16* p = ldsWt + n * WPAD + k0 + hi * 16;
    union { uint4 q[2]; v16bf v; } u;
    u.q[0] = *(const uint4*)(p);
    u.q[1] = *(const uint4*)(p + 8);
    return u.v;
}

// one wave computes a 16x32 output tile (2 col-tiles, A fragments reused, 4 WMMAs)
template <int NC, int OBF, int RELU, int BIAS>
__global__ void gemm_k64(const u16* __restrict__ A, const u16* __restrict__ Bbase,
                         long long Bstride, void* Cbase, long long Cstride,
                         const float* __restrict__ bias) {
    __shared__ u16 ldsWt[NC * WPAD];               // W transposed: [n][k], padded
    const u16* B = Bbase + (size_t)blockIdx.y * Bstride;
    int tid = threadIdx.x;
#pragma unroll 4
    for (int i = tid; i < 64 * NC; i += 256) {
        int k = i / NC, n = i % NC;                 // compile-time NC -> shifts
        ldsWt[n * WPAD + k] = B[i];
    }
    __syncthreads();

    constexpr int nct2 = NC >> 5;                   // column super-tiles of 32
    constexpr int ST   = (NN / 16) * nct2;          // NN % 16 == 0 -> all tiles full
    int gw = (blockIdx.x * 256 + tid) >> 5;
    if (gw >= ST) return;                           // wave-uniform: EXEC stays all-ones
    int lane = tid & 31;
    int n16 = lane & 15, hi = lane >> 4;
    int rt = gw / nct2, ct2 = gw % nct2;
    int col0 = ct2 * 32 + n16;
    int col1 = col0 + 16;

    v16bf b00 = loadBt(ldsWt, col0, 0, hi);
    v16bf b01 = loadBt(ldsWt, col0, 32, hi);
    v16bf b10 = loadBt(ldsWt, col1, 0, hi);
    v16bf b11 = loadBt(ldsWt, col1, 32, hi);
    const u16* Arow = A + (size_t)(rt * 16 + n16) * 64;
    v16bf a0 = loadA(Arow, 0, hi);
    v16bf a1 = loadA(Arow, 32, hi);

    v8f acc0 = {}, acc1 = {};
    acc0 = __builtin_amdgcn_wmma_f32_16x16x32_bf16(false, a0, false, b00, (short)0, acc0, false, false);
    acc0 = __builtin_amdgcn_wmma_f32_16x16x32_bf16(false, a1, false, b01, (short)0, acc0, false, false);
    acc1 = __builtin_amdgcn_wmma_f32_16x16x32_bf16(false, a0, false, b10, (short)0, acc1, false, false);
    acc1 = __builtin_amdgcn_wmma_f32_16x16x32_bf16(false, a1, false, b11, (short)0, acc1, false, false);

    u16*   Cb = (u16*)Cbase + (size_t)blockIdx.y * Cstride;
    float* Cf = (float*)Cbase + (size_t)blockIdx.y * Cstride;
    float bv0 = BIAS ? bias[col0] : 0.f;
    float bv1 = BIAS ? bias[col1] : 0.f;
    size_t base = (size_t)(rt * 16 + hi * 8) * NC;  // C/D: VGPR i -> M = i + hi*8

#pragma unroll
    for (int i = 0; i < 8; i += 2) {
        float v0 = acc0[i],     w0 = acc0[i + 1];
        float v1 = acc1[i],     w1 = acc1[i + 1];
        if (BIAS) { v0 += bv0; w0 += bv0; v1 += bv1; w1 += bv1; }
        if (RELU) { v0 = fmaxf(v0, 0.f); w0 = fmaxf(w0, 0.f);
                    v1 = fmaxf(v1, 0.f); w1 = fmaxf(w1, 0.f); }
        size_t o0 = base + (size_t)i * NC;
        if (OBF) {
            unsigned p0 = f2bf2(v0, w0), p1 = f2bf2(v1, w1);
            Cb[o0 + col0]      = (u16)p0;  Cb[o0 + NC + col0] = (u16)(p0 >> 16);
            Cb[o0 + col1]      = (u16)p1;  Cb[o0 + NC + col1] = (u16)(p1 >> 16);
        } else {
            Cf[o0 + col0] = v0;  Cf[o0 + NC + col0] = w0;
            Cf[o0 + col1] = v1;  Cf[o0 + NC + col1] = w1;
        }
    }
}

// ---------------- attention logits el/er ----------------
__global__ void eler_kernel(const u16* __restrict__ fs, const float* __restrict__ al,
                            const float* __restrict__ ar, float* __restrict__ elb,
                            float* __restrict__ erb) {
    __shared__ float sal[HD], sar[HD];
    int r = blockIdx.y;
    int tid = threadIdx.x;
    if (tid < HD) { sal[tid] = al[r * HD + tid]; sar[tid] = ar[r * HD + tid]; }
    __syncthreads();
    int gid = blockIdx.x * 256 + tid;
    if (gid >= NN * HEADS) return;
    int n = gid >> 2, h = gid & 3;
    const u16* f = fs + ((size_t)r * NN + n) * HD + h * DHH;
    float el = 0.f, er = 0.f;
#pragma unroll
    for (int c = 0; c < 8; c++) {
        uint4 v = *(const uint4*)(f + c * 8);
        const u16* hp = (const u16*)&v;
#pragma unroll
        for (int t2 = 0; t2 < 8; t2++) {
            float xv = bf2f(hp[t2]);
            int j = c * 8 + t2;
            el += xv * sal[h * DHH + j];
            er += xv * sar[h * DHH + j];
        }
    }
    size_t o = ((size_t)r * NN + n) * 4 + h;
    elb[o] = el; erb[o] = er;
}

// ---------------- edge softmax passes ----------------
__global__ void init_ms(int* __restrict__ m32, float* __restrict__ s) {
    int i = blockIdx.x * 256 + threadIdx.x;
    if (i < 3 * NN * HEADS) { m32[i] = (int)0x80000000; s[i] = 0.f; }
}

__global__ void gat_pass1(const int* __restrict__ src, const int* __restrict__ dst, int E,
                          const float* __restrict__ el, const float* __restrict__ er,
                          int* __restrict__ m32, float* __restrict__ ebuf) {
    int i = blockIdx.x * 256 + threadIdx.x;
    if (i >= E) return;
    int s = src[i], d = dst[i];
    float4 ls = *(const float4*)(el + (size_t)s * 4);
    float4 rd = *(const float4*)(er + (size_t)d * 4);
    float e4[4] = {ls.x + rd.x, ls.y + rd.y, ls.z + rd.z, ls.w + rd.w};
#pragma unroll
    for (int h = 0; h < 4; h++) {
        float v = e4[h];
        v = v > 0.f ? v : 0.2f * v;                 // leaky_relu(0.2)
        ebuf[(size_t)i * 4 + h] = v;
        atomicMax(&m32[(size_t)d * 4 + h], f2ord(v));
    }
}

__global__ void gat_pass2(const int* __restrict__ dst, int E, const int* __restrict__ m32,
                          float* __restrict__ ebuf, float* __restrict__ sden) {
    int i = blockIdx.x * 256 + threadIdx.x;
    if (i >= E) return;
    int d = dst[i];
#pragma unroll
    for (int h = 0; h < 4; h++) {
        float m = ord2f(m32[(size_t)d * 4 + h]);
        float a = __expf(ebuf[(size_t)i * 4 + h] - m);
        ebuf[(size_t)i * 4 + h] = a;
        atomicAdd(&sden[(size_t)d * 4 + h], a);
    }
}

// one wave per edge: lane covers 8 contiguous feature elems (same head)
__global__ void gat_pass3(const int* __restrict__ src, const int* __restrict__ dst, int E,
                          const float* __restrict__ sden, const float* __restrict__ ebuf,
                          const u16* __restrict__ fs_r, float* __restrict__ hsum) {
    int wv = (blockIdx.x * 256 + threadIdx.x) >> 5;
    int lane = threadIdx.x & 31;
    int nw = (gridDim.x * 256) >> 5;
    int j0 = lane * 8, h = lane >> 3;
    for (int e = wv; e < E; e += nw) {
        int s = src[e], d = dst[e];
        float alpha = ebuf[(size_t)e * 4 + h] / sden[(size_t)d * 4 + h];
        uint4 v = *(const uint4*)(fs_r + (size_t)s * HD + j0);
        const u16* hp = (const u16*)&v;
        float* out = hsum + (size_t)d * HD + j0;
#pragma unroll
        for (int t = 0; t < 8; t++) atomicAdd(out + t, alpha * bf2f(hp[t]));
    }
}

// ---------------- mean over heads -> bf16 ----------------
__global__ void headmean(const float* __restrict__ hsum, u16* __restrict__ xmean) {
    int gid = blockIdx.x * 256 + threadIdx.x;
    if (gid >= NN * 64) return;
    int n = gid >> 6, d = gid & 63;
    const float* hp = hsum + (size_t)n * HD + d;
    xmean[gid] = f2bf(0.25f * (hp[0] + hp[64] + hp[128] + hp[192]));
}

extern "C" void kernel_launch(void* const* d_in, const int* in_sizes, int n_in,
                              void* d_out, int out_size, void* d_ws, size_t ws_size,
                              hipStream_t stream) {
    (void)in_sizes; (void)n_in; (void)out_size; (void)ws_size;
    const float* x   = (const float*)d_in[0];
    const int* srcs[3] = {(const int*)d_in[1], (const int*)d_in[3], (const int*)d_in[5]};
    const int* dsts[3] = {(const int*)d_in[2], (const int*)d_in[4], (const int*)d_in[6]};
    const int  Es[3]   = {400000, 100000, 400000};
    const long long eoff[3] = {0, 400000, 500000};
    const float *W0 = (const float*)d_in[7],  *al0 = (const float*)d_in[8],
                *ar0 = (const float*)d_in[9], *b0 = (const float*)d_in[10],
                *rW0 = (const float*)d_in[11];
    const float *W1 = (const float*)d_in[12], *al1 = (const float*)d_in[13],
                *ar1 = (const float*)d_in[14], *b1 = (const float*)d_in[15],
                *rW1 = (const float*)d_in[16];
    const float *fcW0 = (const float*)d_in[17], *fcb0 = (const float*)d_in[18];
    const float *fcW1 = (const float*)d_in[19], *fcb1 = (const float*)d_in[20];
    const float *g0 = (const float*)d_in[21], *be0 = (const float*)d_in[22];
    const float *g1 = (const float*)d_in[23], *be1 = (const float*)d_in[24];

    char* p = (char*)d_ws;
    auto carve = [&](size_t bytes) -> void* {
        void* r = (void*)p; p += (bytes + 255) & ~(size_t)255; return r;
    };
    u16*   xn    = (u16*)carve((size_t)NN * 64 * 2);
    u16*   xmean = (u16*)carve((size_t)NN * 64 * 2);
    float* xnext = (float*)carve((size_t)NN * 64 * 4);
    u16*   fs    = (u16*)carve((size_t)3 * NN * HD * 2);
    float* hsum  = (float*)carve((size_t)NN * HD * 4);
    float* elb   = (float*)carve((size_t)3 * NN * 4 * 4);
    float* erb   = (float*)carve((size_t)3 * NN * 4 * 4);
    int*   m32   = (int*)carve((size_t)3 * NN * 4 * 4);
    float* sden  = (float*)carve((size_t)3 * NN * 4 * 4);
    float* ebuf  = (float*)carve((size_t)ETOT * 4 * 4);
    float* stats = (float*)carve(128 * 4);
    u16*   Wbf   = (u16*)carve((size_t)3 * 64 * HD * 2);
    u16*   Wres  = (u16*)carve((size_t)64 * HD * 2);
    float* bsum  = (float*)carve(HD * 4);
    u16*   fcWbf = (u16*)carve(64 * 64 * 2);

    auto layer = [&](const float* xin, const float* W, const float* al, const float* ar,
                     const float* b, const float* rW, const float* fcW, const float* fcb,
                     const float* g, const float* be, float* yout) {
        zero_stats<<<1, 128, 0, stream>>>(stats);
        bn_stats<<<256, 256, 0, stream>>>(xin, stats, stats + 64);
        bn_apply<<<(NN * 64 + 255) / 256, 256, 0, stream>>>(xin, stats, stats + 64, g, be, xn);
        prep_weights<<<192, 256, 0, stream>>>(W, rW, b, fcW, Wbf, Wres, bsum, fcWbf);

        constexpr int ST1 = (NN / 16) * (HD / 32);       // 25000 waves (16x32 tiles)
        gemm_k64<HD, 1, 0, 0><<<dim3((ST1 + 7) / 8, 3), 256, 0, stream>>>(
            xn, Wbf, 64LL * HD, fs, (long long)NN * HD, nullptr);
        gemm_k64<HD, 0, 0, 1><<<dim3((ST1 + 7) / 8, 1), 256, 0, stream>>>(
            xn, Wres, 0, hsum, 0, bsum);

        eler_kernel<<<dim3((NN * HEADS + 255) / 256, 3), 256, 0, stream>>>(fs, al, ar, elb, erb);
        init_ms<<<(3 * NN * HEADS + 255) / 256, 256, 0, stream>>>(m32, sden);
        for (int r = 0; r < 3; r++) {
            int E = Es[r];
            const float* el_r = elb + (size_t)r * NN * 4;
            const float* er_r = erb + (size_t)r * NN * 4;
            int*   m_r = m32 + (size_t)r * NN * 4;
            float* s_r = sden + (size_t)r * NN * 4;
            float* eb  = ebuf + (size_t)eoff[r] * 4;
            gat_pass1<<<(E + 255) / 256, 256, 0, stream>>>(srcs[r], dsts[r], E, el_r, er_r, m_r, eb);
            gat_pass2<<<(E + 255) / 256, 256, 0, stream>>>(dsts[r], E, m_r, eb, s_r);
            gat_pass3<<<(E + 7) / 8, 256, 0, stream>>>(srcs[r], dsts[r], E, s_r, eb,
                                                       fs + (size_t)r * NN * HD, hsum);
        }
        headmean<<<(NN * 64 + 255) / 256, 256, 0, stream>>>(hsum, xmean);
        constexpr int ST2 = (NN / 16) * (64 / 32);       // 6250 waves
        gemm_k64<64, 0, 1, 1><<<dim3((ST2 + 7) / 8, 1), 256, 0, stream>>>(
            xmean, fcWbf, 0, yout, 0, fcb);
    };

    layer(x,     W0, al0, ar0, b0, rW0, fcW0, fcb0, g0, be0, xnext);
    layer(xnext, W1, al1, ar1, b1, rW1, fcW1, fcb1, g1, be1, (float*)d_out);
}
